// Model_86732569575728
// MI455X (gfx1250) — matrix operand order repaired and checked by reading, not compiled, not run on hardware
//
#include <hip/hip_runtime.h>

// ---------------- CDNA5 WMMA types ----------------
typedef __attribute__((ext_vector_type(16))) __bf16 v16bf;
typedef __attribute__((ext_vector_type(8)))  float  v8f;

union BF16Frag {
    v16bf v;
    unsigned int d[8];
};

__device__ __forceinline__ unsigned short f2bf(float f) {
    unsigned int u = __float_as_uint(f);
    unsigned int r = u + 0x7FFFu + ((u >> 16) & 1u);   // round-to-nearest-even
    return (unsigned short)(r >> 16);
}

// Model dims
#define BT   64      // B*T
#define NN   1024    // code vocab
#define GG   128     // GCN width (= H)
#define CSZ  64
#define TAA  256

// ---------------- setup: prior normalize + W2/Wc transpose->bf16 ----------------
__global__ void k_setup(const float* __restrict__ prior,
                        const float* __restrict__ W2,
                        const float* __restrict__ Wc,
                        float* __restrict__ pn,
                        unsigned short* __restrict__ w2t,
                        unsigned short* __restrict__ wct) {
    int tid = threadIdx.x;                   // 1024 threads
    __shared__ float red[1024];
    red[tid] = prior[tid];
    __syncthreads();
    for (int s = 512; s > 0; s >>= 1) {
        if (tid < s) red[tid] += red[tid + s];
        __syncthreads();
    }
    pn[tid] = prior[tid] / red[0];
    for (int idx = tid; idx < GG * GG; idx += 1024) {
        int i = idx >> 7, j = idx & 127;     // W[i][j] -> Wt[j][i]
        w2t[j * GG + i] = f2bf(W2[idx]);
        wct[j * GG + i] = f2bf(Wc[idx]);
    }
}

// ---------------- EW1 = c_emb @ W1  (fp32, tiny) ----------------
__global__ void k_ew1(const float* __restrict__ c_emb,
                      const float* __restrict__ W1,
                      float* __restrict__ EW1) {
    int t = blockIdx.x * blockDim.x + threadIdx.x;   // 1024*128
    int n = t >> 7, g = t & 127;
    float s = 0.f;
    #pragma unroll 8
    for (int k = 0; k < CSZ; ++k) s += c_emb[n * CSZ + k] * W1[k * GG + g];
    EW1[t] = s;
}

// ---------------- adj -> bf16 ----------------
__global__ void k_adjbf(const float* __restrict__ adj, unsigned short* __restrict__ adjbf) {
    int t = blockIdx.x * blockDim.x + threadIdx.x;   // 1024*1024
    adjbf[t] = f2bf(adj[t]);
}

// ---------------- x0^T (feature-major) bf16: x0t[bt][g][n] = code_x[bt][n]*EW1[n][g] ----------------
__global__ void k_x0t(const float* __restrict__ code_x,
                      const float* __restrict__ EW1,
                      unsigned short* __restrict__ x0t) {
    int t = blockIdx.x * blockDim.x + threadIdx.x;   // 64*128*1024
    int n  = t & 1023;
    int g  = (t >> 10) & 127;
    int bt = t >> 17;
    x0t[t] = f2bf(code_x[bt * NN + n] * EW1[n * GG + g]);
}

// =====================================================================
// Big adj GEMMs: one wave computes a 32x128 strip (2 M-tiles x 8 F-tiles)
// B fragments are double-buffered and reused by two WMMAs each.
// __launch_bounds__(32,1): allow ~180 VGPRs (no spills), 1 wave/block.
// =====================================================================

// ---------------- GEMM1: h1 = relu(adj @ x0 + b1), bf16 out [bt][n][g] ----------------
__global__ void __launch_bounds__(32, 1)
k_gcn_adj1(const unsigned short* __restrict__ adjbf,
           const unsigned short* __restrict__ x0t,   // [bt][g][n]
           const float* __restrict__ b1,
           unsigned short* __restrict__ h1bf) {      // [bt][n][g]
    const int mt = blockIdx.x, bt = blockIdx.y;      // mt: 0..31
    const int lane = threadIdx.x;
    const int lm = lane & 15, lh = lane >> 4;
    const int m0 = mt * 32;
    const unsigned short* __restrict__ arow0 = adjbf + (size_t)(m0 + lm) * NN;
    const unsigned short* __restrict__ arow1 = adjbf + (size_t)(m0 + 16 + lm) * NN;
    const unsigned short* __restrict__ Xb    = x0t + (size_t)bt * GG * NN;

    v8f zero = {0.f,0.f,0.f,0.f,0.f,0.f,0.f,0.f};
    v8f acc0[8], acc1[8];
    #pragma unroll
    for (int j = 0; j < 8; ++j) { acc0[j] = zero; acc1[j] = zero; }

    for (int k0 = 0; k0 < NN; k0 += 32) {
        BF16Frag a0, a1;
        #pragma unroll
        for (int v = 0; v < 8; ++v) {
            int k = k0 + lh * 16 + v * 2;
            a0.d[v] = *(const unsigned int*)(arow0 + k);
            a1.d[v] = *(const unsigned int*)(arow1 + k);
        }
        if (k0 + 32 < NN) {                 // warm next adj block (global_prefetch)
            __builtin_prefetch(arow0 + k0 + 32, 0, 0);
            __builtin_prefetch(arow1 + k0 + 32, 0, 0);
        }
        BF16Frag b[2];
        {
            const unsigned short* xc = Xb + (size_t)lm * NN + k0 + lh * 16;
            #pragma unroll
            for (int v = 0; v < 8; ++v) b[0].d[v] = *(const unsigned int*)(xc + v * 2);
        }
        #pragma unroll
        for (int j = 0; j < 8; ++j) {
            if (j < 7) {                    // issue next B before consuming current
                const unsigned short* xc = Xb + (size_t)((j + 1) * 16 + lm) * NN + k0 + lh * 16;
                #pragma unroll
                for (int v = 0; v < 8; ++v) b[(j + 1) & 1].d[v] = *(const unsigned int*)(xc + v * 2);
            }
            acc0[j] = __builtin_amdgcn_wmma_f32_16x16x32_bf16(
                false, a0.v, false, b[j & 1].v, (short)0, acc0[j], false, false);
            acc1[j] = __builtin_amdgcn_wmma_f32_16x16x32_bf16(
                false, a1.v, false, b[j & 1].v, (short)0, acc1[j], false, false);
        }
    }
    unsigned short* __restrict__ H = h1bf + (size_t)bt * NN * GG;
    #pragma unroll
    for (int j = 0; j < 8; ++j) {
        float bb = b1[j * 16 + lm];
        #pragma unroll
        for (int v = 0; v < 8; ++v) {
            float x0v = acc0[j][v] + bb;
            float x1v = acc1[j][v] + bb;
            x0v = x0v > 0.f ? x0v : 0.f;
            x1v = x1v > 0.f ? x1v : 0.f;
            H[(size_t)(m0 + lh * 8 + v) * GG + j * 16 + lm]      = f2bf(x0v);
            H[(size_t)(m0 + 16 + lh * 8 + v) * GG + j * 16 + lm] = f2bf(x1v);
        }
    }
}

// ---------------- GEMM3: co = adj @ y + b2, fp32 out [bt][n][g] ----------------
__global__ void __launch_bounds__(32, 1)
k_gcn_adj2(const unsigned short* __restrict__ adjbf,
           const unsigned short* __restrict__ yt,    // [bt][g][n]
           const float* __restrict__ b2,
           float* __restrict__ co) {                 // [bt][n][g]
    const int mt = blockIdx.x, bt = blockIdx.y;      // mt: 0..31
    const int lane = threadIdx.x;
    const int lm = lane & 15, lh = lane >> 4;
    const int m0 = mt * 32;
    const unsigned short* __restrict__ arow0 = adjbf + (size_t)(m0 + lm) * NN;
    const unsigned short* __restrict__ arow1 = adjbf + (size_t)(m0 + 16 + lm) * NN;
    const unsigned short* __restrict__ Yb    = yt + (size_t)bt * GG * NN;

    v8f zero = {0.f,0.f,0.f,0.f,0.f,0.f,0.f,0.f};
    v8f acc0[8], acc1[8];
    #pragma unroll
    for (int j = 0; j < 8; ++j) { acc0[j] = zero; acc1[j] = zero; }

    for (int k0 = 0; k0 < NN; k0 += 32) {
        BF16Frag a0, a1;
        #pragma unroll
        for (int v = 0; v < 8; ++v) {
            int k = k0 + lh * 16 + v * 2;
            a0.d[v] = *(const unsigned int*)(arow0 + k);
            a1.d[v] = *(const unsigned int*)(arow1 + k);
        }
        if (k0 + 32 < NN) {
            __builtin_prefetch(arow0 + k0 + 32, 0, 0);
            __builtin_prefetch(arow1 + k0 + 32, 0, 0);
        }
        BF16Frag b[2];
        {
            const unsigned short* yc = Yb + (size_t)lm * NN + k0 + lh * 16;
            #pragma unroll
            for (int v = 0; v < 8; ++v) b[0].d[v] = *(const unsigned int*)(yc + v * 2);
        }
        #pragma unroll
        for (int j = 0; j < 8; ++j) {
            if (j < 7) {
                const unsigned short* yc = Yb + (size_t)((j + 1) * 16 + lm) * NN + k0 + lh * 16;
                #pragma unroll
                for (int v = 0; v < 8; ++v) b[(j + 1) & 1].d[v] = *(const unsigned int*)(yc + v * 2);
            }
            acc0[j] = __builtin_amdgcn_wmma_f32_16x16x32_bf16(
                false, a0.v, false, b[j & 1].v, (short)0, acc0[j], false, false);
            acc1[j] = __builtin_amdgcn_wmma_f32_16x16x32_bf16(
                false, a1.v, false, b[j & 1].v, (short)0, acc1[j], false, false);
        }
    }
    float* __restrict__ C = co + (size_t)bt * NN * GG;
    #pragma unroll
    for (int j = 0; j < 8; ++j) {
        float bb = b2[j * 16 + lm];
        #pragma unroll
        for (int v = 0; v < 8; ++v) {
            C[(size_t)(m0 + lh * 8 + v) * GG + j * 16 + lm]      = acc0[j][v] + bb;
            C[(size_t)(m0 + 16 + lh * 8 + v) * GG + j * 16 + lm] = acc1[j][v] + bb;
        }
    }
}

// ---------------- GEMM2: y = h1 @ W2, store transposed bf16 yt[bt][g][n] ----------------
__global__ void k_gcn_w2(const unsigned short* __restrict__ h1bf,  // [bt][n][g]
                         const unsigned short* __restrict__ w2t,   // [gout][gin]
                         unsigned short* __restrict__ yt) {        // [bt][gout][n]
    const int mt = blockIdx.x, bt = blockIdx.y;
    const int lane = threadIdx.x;
    const int lm = lane & 15, lh = lane >> 4;
    const int m0 = mt * 16;
    const unsigned short* __restrict__ A = h1bf + (size_t)bt * NN * GG + (size_t)(m0 + lm) * GG;

    v8f zero = {0.f,0.f,0.f,0.f,0.f,0.f,0.f,0.f};
    v8f acc[8];
    #pragma unroll
    for (int j = 0; j < 8; ++j) acc[j] = zero;

    #pragma unroll
    for (int k0 = 0; k0 < GG; k0 += 32) {
        BF16Frag a;
        #pragma unroll
        for (int v = 0; v < 8; ++v) {
            int k = k0 + lh * 16 + v * 2;
            a.d[v] = *(const unsigned int*)(A + k);
        }
        #pragma unroll
        for (int j = 0; j < 8; ++j) {
            BF16Frag b;
            const unsigned short* wc = w2t + (size_t)(j * 16 + lm) * GG + k0 + lh * 16;
            #pragma unroll
            for (int v = 0; v < 8; ++v) b.d[v] = *(const unsigned int*)(wc + v * 2);
            acc[j] = __builtin_amdgcn_wmma_f32_16x16x32_bf16(
                false, a.v, false, b.v, (short)0, acc[j], false, false);
        }
    }
    unsigned short* __restrict__ Y = yt + (size_t)bt * GG * NN;
    #pragma unroll
    for (int j = 0; j < 8; ++j)
        #pragma unroll
        for (int v = 0; v < 8; ++v)
            Y[(size_t)(j * 16 + lm) * NN + m0 + lh * 8 + v] = f2bf(acc[j][v]);
}

// ---------------- prior modulation + layernorm + leakyrelu + residual (in-place) ----------------
__global__ void k_mod_ln(float* __restrict__ co,
                         const float* __restrict__ pn,
                         const float* __restrict__ Wp,
                         const float* __restrict__ bp,
                         const float* __restrict__ ln_g,
                         const float* __restrict__ ln_b) {
    int r = blockIdx.x;               // bt*1024 + n
    int g = threadIdx.x;              // 128
    int n = r & 1023;
    __shared__ float red[GG];
    float c  = co[(size_t)r * GG + g];
    float po = pn[n] * Wp[g] + bp[g];
    float x  = c * po;
    red[g] = x; __syncthreads();
    for (int s = 64; s > 0; s >>= 1) { if (g < s) red[g] += red[g + s]; __syncthreads(); }
    float mu = red[0] * (1.f / GG); __syncthreads();
    float d = x - mu;
    red[g] = d * d; __syncthreads();
    for (int s = 64; s > 0; s >>= 1) { if (g < s) red[g] += red[g + s]; __syncthreads(); }
    float var = red[0] * (1.f / GG);
    float xn = d * rsqrtf(var + 1e-5f) * ln_g[g] + ln_b[g];
    xn = xn >= 0.f ? xn : 0.01f * xn;
    co[(size_t)r * GG + g] = c + xn;
}

// ---------------- logits = tanh(co@Wc+bc) . vc   (WMMA + shuffle reduce) ----------------
__global__ void k_attn_logits(const float* __restrict__ co,         // [bt][n][g]
                              const unsigned short* __restrict__ wct,
                              const float* __restrict__ bc,
                              const float* __restrict__ vc,
                              float* __restrict__ logits) {         // [bt][n]
    const int mt = blockIdx.x, bt = blockIdx.y;
    const int lane = threadIdx.x;
    const int lm = lane & 15, lh = lane >> 4;
    const int m0 = mt * 16;
    const float* __restrict__ A = co + (size_t)bt * NN * GG + (size_t)(m0 + lm) * GG;

    v8f zero = {0.f,0.f,0.f,0.f,0.f,0.f,0.f,0.f};
    v8f acc[8];
    #pragma unroll
    for (int j = 0; j < 8; ++j) acc[j] = zero;

    #pragma unroll
    for (int k0 = 0; k0 < GG; k0 += 32) {
        BF16Frag a;
        #pragma unroll
        for (int v = 0; v < 8; ++v) {
            int k = k0 + lh * 16 + v * 2;
            unsigned int lo = f2bf(A[k]);
            unsigned int hi = f2bf(A[k + 1]);
            a.d[v] = lo | (hi << 16);
        }
        #pragma unroll
        for (int j = 0; j < 8; ++j) {
            BF16Frag b;
            const unsigned short* wc = wct + (size_t)(j * 16 + lm) * GG + k0 + lh * 16;
            #pragma unroll
            for (int v = 0; v < 8; ++v) b.d[v] = *(const unsigned int*)(wc + v * 2);
            acc[j] = __builtin_amdgcn_wmma_f32_16x16x32_bf16(
                false, a.v, false, b.v, (short)0, acc[j], false, false);
        }
    }
    float vcv[8], bcv[8];
    #pragma unroll
    for (int j = 0; j < 8; ++j) { vcv[j] = vc[j * 16 + lm]; bcv[j] = bc[j * 16 + lm]; }
    float* __restrict__ L = logits + (size_t)bt * NN;
    #pragma unroll
    for (int v = 0; v < 8; ++v) {
        float s = 0.f;
        #pragma unroll
        for (int j = 0; j < 8; ++j) s += vcv[j] * tanhf(acc[j][v] + bcv[j]);
        // reduce across the 16 column lanes within each half-wave
        #pragma unroll
        for (int off = 1; off < 16; off <<= 1) s += __shfl_xor(s, off, 32);
        if (lm == 0) L[m0 + lh * 8 + v] = s;
    }
}

// ---------------- masked softmax over codes + visit = attn @ co ----------------
__global__ void k_softmax_visit(const float* __restrict__ logits,
                                const float* __restrict__ code_x,
                                const float* __restrict__ co,
                                float* __restrict__ visit) {        // [bt][g]
    int bt = blockIdx.x, tid = threadIdx.x;                         // 256 threads
    __shared__ float attn[NN];
    __shared__ float red[256];
    const float* __restrict__ L  = logits + (size_t)bt * NN;
    const float* __restrict__ CX = code_x + (size_t)bt * NN;
    float lv[4], lmax = -1e30f;
    #pragma unroll
    for (int i = 0; i < 4; ++i) {
        int n = tid + i * 256;
        float l = (CX[n] != 0.f) ? L[n] : -1e9f;
        lv[i] = l; lmax = fmaxf(lmax, l);
    }
    red[tid] = lmax; __syncthreads();
    for (int s = 128; s > 0; s >>= 1) { if (tid < s) red[tid] = fmaxf(red[tid], red[tid + s]); __syncthreads(); }
    float mx = red[0]; __syncthreads();
    float ps = 0.f;
    #pragma unroll
    for (int i = 0; i < 4; ++i) { float e = expf(lv[i] - mx); attn[tid + i * 256] = e; ps += e; }
    red[tid] = ps; __syncthreads();
    for (int s = 128; s > 0; s >>= 1) { if (tid < s) red[tid] += red[tid + s]; __syncthreads(); }
    float inv = 1.f / red[0]; __syncthreads();
    #pragma unroll
    for (int i = 0; i < 4; ++i) attn[tid + i * 256] *= inv;
    __syncthreads();
    int g = tid & 127, h = tid >> 7;
    const float* __restrict__ C = co + (size_t)bt * NN * GG;
    float a = 0.f;
    for (int n = h * 512; n < h * 512 + 512; ++n) a += attn[n] * C[(size_t)n * GG + g];
    red[tid] = a; __syncthreads();
    if (h == 0) visit[bt * GG + g] = red[g] + red[128 + g];
}

// ---------------- LSTM over T + temporal attention + classifier ----------------
__global__ void k_lstm_head(const float* __restrict__ visit,   // [B][T][G]
                            const float* __restrict__ Wih, const float* __restrict__ Whh,
                            const float* __restrict__ bih, const float* __restrict__ bhh,
                            const float* __restrict__ Wt,  const float* __restrict__ btv,
                            const float* __restrict__ vt,
                            const float* __restrict__ Wcls, const float* __restrict__ bcls,
                            float* __restrict__ out) {
    int b = blockIdx.x, tid = threadIdx.x;                     // 512 threads
    __shared__ float h[GG], c[GG], z[4 * GG], hs[8][GG], red[512], tw[8];
    if (tid < GG) { h[tid] = 0.f; c[tid] = 0.f; }
    __syncthreads();
    for (int t = 0; t < 8; ++t) {
        const float* x = visit + (size_t)(b * 8 + t) * GG;
        float zz = bih[tid] + bhh[tid];
        for (int k = 0; k < GG; ++k)
            zz += x[k] * Wih[k * 512 + tid] + h[k] * Whh[k * 512 + tid];
        z[tid] = zz; __syncthreads();
        if (tid < GG) {
            float ig = 1.f / (1.f + expf(-z[tid]));
            float fg = 1.f / (1.f + expf(-z[GG + tid]));
            float gg = tanhf(z[2 * GG + tid]);
            float og = 1.f / (1.f + expf(-z[3 * GG + tid]));
            float cn = fg * c[tid] + ig * gg;
            c[tid] = cn;
            float hn = og * tanhf(cn);
            h[tid] = hn;
            hs[t][tid] = hn;
        }
        __syncthreads();
    }
    for (int t = 0; t < 8; ++t) {
        float u = 0.f;
        if (tid < TAA) {
            float s = btv[tid];
            for (int k = 0; k < GG; ++k) s += hs[t][k] * Wt[k * TAA + tid];
            u = tanhf(s) * vt[tid];
        }
        red[tid] = (tid < TAA) ? u : 0.f; __syncthreads();
        for (int s = 256; s > 0; s >>= 1) { if (tid < s) red[tid] += red[tid + s]; __syncthreads(); }
        if (tid == 0) tw[t] = red[0];
        __syncthreads();
    }
    if (tid == 0) {
        float mx = -1e30f;
        for (int t = 0; t < 8; ++t) mx = fmaxf(mx, tw[t]);
        float s = 0.f;
        for (int t = 0; t < 8; ++t) { tw[t] = expf(tw[t] - mx); s += tw[t]; }
        for (int t = 0; t < 8; ++t) tw[t] /= s;
    }
    __syncthreads();
    float p = 0.f;
    if (tid < GG) {
        for (int t = 0; t < 8; ++t) p += tw[t] * hs[t][tid];
        p *= Wcls[tid];
    }
    red[tid] = (tid < GG) ? p : 0.f; __syncthreads();
    for (int s = 256; s > 0; s >>= 1) { if (tid < s) red[tid] += red[tid + s]; __syncthreads(); }
    if (tid == 0) out[b] = red[0] + bcls[0];
}

// ---------------- host launcher ----------------
extern "C" void kernel_launch(void* const* d_in, const int* in_sizes, int n_in,
                              void* d_out, int out_size, void* d_ws, size_t ws_size,
                              hipStream_t stream) {
    const float* code_x = (const float*)d_in[0];
    const float* prior  = (const float*)d_in[4];
    const float* adj    = (const float*)d_in[5];
    const float* c_emb  = (const float*)d_in[6];
    const float* W1     = (const float*)d_in[7];
    const float* b1     = (const float*)d_in[8];
    const float* W2     = (const float*)d_in[9];
    const float* b2     = (const float*)d_in[10];
    const float* Wp     = (const float*)d_in[11];
    const float* bp     = (const float*)d_in[12];
    const float* ln_g   = (const float*)d_in[13];
    const float* ln_b   = (const float*)d_in[14];
    const float* Wc     = (const float*)d_in[15];
    const float* bc     = (const float*)d_in[16];
    const float* vc     = (const float*)d_in[17];
    const float* Wt     = (const float*)d_in[18];
    const float* btv    = (const float*)d_in[19];
    const float* vt     = (const float*)d_in[20];
    const float* Wih    = (const float*)d_in[21];
    const float* Whh    = (const float*)d_in[22];
    const float* bih    = (const float*)d_in[23];
    const float* bhh    = (const float*)d_in[24];
    const float* Wcls   = (const float*)d_in[25];
    const float* bcls   = (const float*)d_in[26];
    float* out = (float*)d_out;

    char* ws = (char*)d_ws;
    unsigned short* adjbf = (unsigned short*)(ws + 0);                 // 2 MB
    float*          EW1   = (float*)         (ws + 2097152);          // 512 KB
    float*          pn    = (float*)         (ws + 2621440);          // 4 KB
    unsigned short* w2t   = (unsigned short*)(ws + 2625536);          // 32 KB
    unsigned short* wct   = (unsigned short*)(ws + 2658304);          // 32 KB
    unsigned short* x0t   = (unsigned short*)(ws + 2691072);          // 16 MB (aliased as yt)
    unsigned short* yt    = x0t;
    unsigned short* h1bf  = (unsigned short*)(ws + 19468288);         // 16 MB
    float*          co    = (float*)         (ws + 36245504);         // 32 MB
    float*          logit = (float*)         (ws + 69799936);         // 256 KB
    float*          vis   = (float*)         (ws + 70062080);         // 32 KB

    k_setup<<<1, 1024, 0, stream>>>(prior, W2, Wc, pn, w2t, wct);
    k_ew1<<<512, 256, 0, stream>>>(c_emb, W1, EW1);
    k_adjbf<<<4096, 256, 0, stream>>>(adj, adjbf);
    k_x0t<<<32768, 256, 0, stream>>>(code_x, EW1, x0t);

    dim3 adjGrid(32, 64);        // 2 M-tiles per wave
    dim3 smallGrid(64, 64);
    k_gcn_adj1<<<adjGrid, 32, 0, stream>>>(adjbf, x0t, b1, h1bf);
    k_gcn_w2  <<<smallGrid, 32, 0, stream>>>(h1bf, w2t, yt);
    k_gcn_adj2<<<adjGrid, 32, 0, stream>>>(adjbf, yt, b2, co);

    k_mod_ln<<<BT * NN, GG, 0, stream>>>(co, pn, Wp, bp, ln_g, ln_b);
    k_attn_logits<<<smallGrid, 32, 0, stream>>>(co, wct, bc, vc, logit);
    k_softmax_visit<<<BT, 256, 0, stream>>>(logit, code_x, co, vis);
    k_lstm_head<<<8, 512, 0, stream>>>(vis, Wih, Whh, bih, bhh, Wt, btv, vt, Wcls, bcls, out);
}
